// LinearAttention_40587440947492
// MI455X (gfx1250) — matrix operand order repaired and checked by reading, not compiled, not run on hardware
//
#include <hip/hip_runtime.h>

// ---------------------------------------------------------------------------
// MoE (B=8,F=2048,S=2048,E=16,cap=1024) for MI455X / gfx1250.
// Pipeline: gates+softmax (f32) -> per-expert bitonic argsort (exact top_k
// order) -> greedy masked selection (perm, gate scale) -> stage Xs (f16,
// shuffle+scale folded) -> stage Wt (f16, transposed) -> WMMA f16 GEMM
// (v_wmma_f32_16x16x32_f16) -> un-permute transpose to (B,F,S).
// ---------------------------------------------------------------------------

typedef _Float16 half_t;
typedef _Float16 v8h  __attribute__((ext_vector_type(8)));
typedef _Float16 v16h __attribute__((ext_vector_type(16)));
typedef float    v8f  __attribute__((ext_vector_type(8)));
typedef float    v4f  __attribute__((ext_vector_type(4)));
typedef unsigned v4u  __attribute__((ext_vector_type(4)));

#define BB   8
#define FF   2048
#define SS   2048
#define EE   16
#define CAP  1024
#define TOK  16384   // BB*SS

__device__ __forceinline__ unsigned pack2h(float a, float b) {
  _Float16 ha = (_Float16)a, hb = (_Float16)b;
  unsigned short ua = __builtin_bit_cast(unsigned short, ha);
  unsigned short ub = __builtin_bit_cast(unsigned short, hb);
  return (unsigned)ua | ((unsigned)ub << 16);
}

// ---------------------------------------------------------------------------
// K1: logits = x @ gate, softmax -> gatesT[e][t]   (x[t,f] = inp[b,f,s])
// 64 blocks x 256 threads, 1 token per thread; gate tiles staged in LDS.
// ---------------------------------------------------------------------------
__global__ __launch_bounds__(256) void k_gates(const float* __restrict__ inp,
                                               const float* __restrict__ gate,
                                               float* __restrict__ gatesT) {
  __shared__ float gld[128 * 16];
  const int tid = threadIdx.x;
  const int t = blockIdx.x * 256 + tid;
  const int b = t >> 11;          // t / SS
  const int s = t & (SS - 1);
  const float* src = inp + (size_t)b * FF * SS + s;

  float acc[16];
#pragma unroll
  for (int e = 0; e < 16; ++e) acc[e] = 0.f;

  for (int f0 = 0; f0 < FF; f0 += 128) {
    __syncthreads();
#pragma unroll
    for (int i = 0; i < 8; ++i) gld[tid + i * 256] = gate[f0 * 16 + tid + i * 256];
    __syncthreads();
    for (int ff = 0; ff < 128; ++ff) {
      float v = src[(size_t)(f0 + ff) * SS];
      const float* g = &gld[ff * 16];
#pragma unroll
      for (int e = 0; e < 16; ++e) acc[e] = fmaf(v, g[e], acc[e]);
    }
  }
  float m = acc[0];
#pragma unroll
  for (int e = 1; e < 16; ++e) m = fmaxf(m, acc[e]);
  float ex[16], sum = 0.f;
#pragma unroll
  for (int e = 0; e < 16; ++e) { ex[e] = __expf(acc[e] - m); sum += ex[e]; }
  float rs = 1.0f / sum;
#pragma unroll
  for (int e = 0; e < 16; ++e) gatesT[e * TOK + t] = ex[e] * rs;
}

// ---------------------------------------------------------------------------
// K2: per-expert full bitonic argsort (descending value, ascending index on
// ties == jax.lax.top_k order). key = ((~bits(val)) << 32) | idx, ascending.
// One block per expert, keys live in global scratch (single-block sync ok).
// ---------------------------------------------------------------------------
__global__ __launch_bounds__(1024) void k_sort(const float* __restrict__ gatesT,
                                               unsigned long long* __restrict__ keys) {
  const int e = blockIdx.x;
  unsigned long long* k = keys + (size_t)e * TOK;
  for (int i = threadIdx.x; i < TOK; i += 1024) {
    unsigned u = __float_as_uint(gatesT[e * TOK + i]);   // vals >= 0: monotonic
    k[i] = ((unsigned long long)(0xFFFFFFFFu - u) << 32) | (unsigned)i;
  }
  __syncthreads();
  for (int size = 2; size <= TOK; size <<= 1) {
    for (int stride = size >> 1; stride > 0; stride >>= 1) {
      for (int i = threadIdx.x; i < TOK; i += 1024) {
        int j = i ^ stride;
        if (j > i) {
          bool up = ((i & size) == 0);
          unsigned long long a = k[i], b = k[j];
          if ((a > b) == up) { k[i] = b; k[j] = a; }
        }
      }
      __syncthreads();
    }
  }
}

// ---------------------------------------------------------------------------
// K3: sequential greedy selection over experts (exact reference semantics).
// perm[c*E+e] = c-th unmasked token in expert e's sorted order.
// scale[t] = gates[t, p//cap] with p = position of t in perm (reference quirk).
// Single block, 1024 threads, masked-prefix-sum per expert.
// ---------------------------------------------------------------------------
__global__ __launch_bounds__(1024) void k_select(const unsigned long long* __restrict__ keys,
                                                 const float* __restrict__ gatesT,
                                                 unsigned* __restrict__ perm,
                                                 float* __restrict__ scale) {
  __shared__ unsigned char taken[TOK];   // 16 KB
  __shared__ unsigned sdata[1024];
  const int tid = threadIdx.x;
  for (int i = tid; i < TOK; i += 1024) taken[i] = 0;
  __syncthreads();

  for (int e = 0; e < EE; ++e) {
    unsigned tl[16]; unsigned fl = 0, cnt = 0;
#pragma unroll
    for (int q = 0; q < 16; ++q) {
      unsigned t = (unsigned)(keys[(size_t)e * TOK + tid * 16 + q] & 0xFFFFFFFFu);
      tl[q] = t;
      unsigned f = taken[t] ? 0u : 1u;
      fl |= (f << q); cnt += f;
    }
    sdata[tid] = cnt;
    __syncthreads();
    unsigned incl = cnt;
    for (int off = 1; off < 1024; off <<= 1) {
      unsigned v = (tid >= off) ? sdata[tid - off] : 0u;
      __syncthreads();
      incl += v; sdata[tid] = incl;
      __syncthreads();
    }
    unsigned c = incl - cnt;   // exclusive prefix
#pragma unroll
    for (int q = 0; q < 16; ++q) {
      if ((fl >> q) & 1u) {
        if (c < CAP) {
          unsigned t = tl[q];
          unsigned p = c * EE + e;
          perm[p] = t;
          scale[t] = gatesT[(p >> 10) * TOK + t];   // expert_index = p // cap
          taken[t] = 1;
        }
        ++c;
      }
    }
    __syncthreads();
  }
}

// ---------------------------------------------------------------------------
// K4: Xs[t][j] = (f16)(inp[b, fs[j], s] * scale[t])   (token-major, 4KB rows)
// LDS transpose tile 32(j) x 64(s): coalesced reads and 64B row-chunk writes.
// ---------------------------------------------------------------------------
__global__ __launch_bounds__(256) void k_xs(const float* __restrict__ inp,
                                            const int* __restrict__ fs,
                                            const float* __restrict__ scale,
                                            half_t* __restrict__ Xs) {
  __shared__ float tile[64 * 33];   // [ss][jj], pitch 33
  __shared__ int fsl[32];
  const int b = blockIdx.z, st = blockIdx.y, jt = blockIdx.x;
  const int s0 = st * 64, j0 = jt * 32;
  const int tid = threadIdx.x;
  if (tid < 32) fsl[tid] = fs[j0 + tid];
  __syncthreads();
#pragma unroll
  for (int p = 0; p < 8; ++p) {
    int idx = p * 256 + tid;
    int jj = idx >> 6, ss = idx & 63;
    tile[ss * 33 + jj] = inp[((size_t)b * FF + fsl[jj]) * SS + s0 + ss];
  }
  __syncthreads();
#pragma unroll
  for (int p = 0; p < 4; ++p) {
    int idx = p * 256 + tid;
    int ss = idx >> 4, jp = idx & 15;
    int t = b * SS + s0 + ss;
    float sc = scale[t];
    unsigned u = pack2h(tile[ss * 33 + 2 * jp] * sc, tile[ss * 33 + 2 * jp + 1] * sc);
    *(unsigned*)(Xs + (size_t)t * FF + j0 + 2 * jp) = u;
  }
}

// ---------------------------------------------------------------------------
// K5: Wt[e][n][k] = (f16)W[e][k][n]  -- one-time transpose+convert so GEMM
// B-operand rows are K-contiguous (both WMMA operands want K-contiguous).
// ---------------------------------------------------------------------------
__global__ __launch_bounds__(256) void k_wt(const float* __restrict__ W,
                                            half_t* __restrict__ Wt) {
  __shared__ float tile[64 * 65];   // [k][n], pitch 65
  const int e = blockIdx.z, kt = blockIdx.y, nt = blockIdx.x;
  const int k0 = kt * 64, n0 = nt * 64;
  const int tid = threadIdx.x;
  const float* src = W + ((size_t)e * FF + k0) * FF + n0;
#pragma unroll
  for (int p = 0; p < 4; ++p) {
    int idx = p * 256 + tid;
    int kk = idx >> 4, nq = idx & 15;
    v4f v = *(const v4f*)(src + (size_t)kk * FF + nq * 4);
#pragma unroll
    for (int j = 0; j < 4; ++j) tile[kk * 65 + nq * 4 + j] = v[j];
  }
  __syncthreads();
  half_t* dst = Wt + ((size_t)e * FF + n0) * FF + k0;
#pragma unroll
  for (int p = 0; p < 8; ++p) {
    int idx = p * 256 + tid;
    int nn = idx >> 5, kp = idx & 31;
    unsigned u = pack2h(tile[(2 * kp) * 65 + nn], tile[(2 * kp + 1) * 65 + nn]);
    *(unsigned*)(dst + (size_t)nn * FF + 2 * kp) = u;
  }
}

// ---------------------------------------------------------------------------
// K6: per-expert GEMM: Y[t] = Xs[t] @ Wt[e]^T-layout, rows gathered by perm.
// Block 128 thr (4 waves), tile 128x128x32, wave tile 64x64 => 16 WMMAs/wave
// per k-step. LDS pitch 40 halfs (16B-aligned frag chunks, even 2x bank load).
// Fragment layouts per CDNA5 ISA 7.12.2:
//   A (16x32 f16): lanes0-15 row M=lane, chunks K[0..7],K[16..23];
//                  lanes16-31 same rows, K[8..15],K[24..31].
//   B (32x16 f16): lanes0-15 col N=lane, K[0..7],K[16..23];
//                  lanes16-31 K[8..15],K[24..31].
// ---------------------------------------------------------------------------
#define BM 128
#define BN 128
#define BK 32
#define LDP 40

__global__ __launch_bounds__(128, 1) void k_gemm(const half_t* __restrict__ Xs,
                                                 const half_t* __restrict__ Wt,
                                                 const unsigned* __restrict__ perm,
                                                 float* __restrict__ Y) {
  __shared__ half_t Alds[BM * LDP];
  __shared__ half_t Blds[BN * LDP];
  __shared__ unsigned tmap[BM];
  const int tid = threadIdx.x;
  const int e = blockIdx.z, mt = blockIdx.y, nt = blockIdx.x;

  tmap[tid] = perm[(mt * BM + tid) * EE + e];          // c -> token (p = c*E+e)
  __syncthreads();

  const int lane = tid & 31, wave = tid >> 5;
  const int wm = (wave >> 1) * 64, wn = (wave & 1) * 64;

  // staging: 512 16B chunks each for A and B, 4 per thread
  const half_t* aptr[4]; const half_t* bptr[4]; int soff[4];
#pragma unroll
  for (int p = 0; p < 4; ++p) {
    int idx = p * 128 + tid;
    int row = idx >> 2, ch = idx & 3;
    soff[p] = row * LDP + ch * 8;
    aptr[p] = Xs + (size_t)tmap[row] * FF + ch * 8;
    bptr[p] = Wt + ((size_t)e * FF + nt * BN + row) * FF + ch * 8;
  }

  v8f acc[4][4];
  const v8f vzero = {0.f, 0.f, 0.f, 0.f, 0.f, 0.f, 0.f, 0.f};
#pragma unroll
  for (int mi = 0; mi < 4; ++mi)
#pragma unroll
    for (int ni = 0; ni < 4; ++ni) acc[mi][ni] = vzero;

  v4u ra[4], rb[4];
#pragma unroll
  for (int p = 0; p < 4; ++p) { ra[p] = *(const v4u*)aptr[p]; rb[p] = *(const v4u*)bptr[p]; }

  const int r = lane & 15, hk = lane >> 4;
  for (int kt = 0; kt < FF / BK; ++kt) {
    __syncthreads();
#pragma unroll
    for (int p = 0; p < 4; ++p) {
      *(v4u*)&Alds[soff[p]] = ra[p];
      *(v4u*)&Blds[soff[p]] = rb[p];
    }
    __syncthreads();
    if (kt < FF / BK - 1) {
#pragma unroll
      for (int p = 0; p < 4; ++p) {
        ra[p] = *(const v4u*)(aptr[p] + (size_t)(kt + 1) * BK);
        rb[p] = *(const v4u*)(bptr[p] + (size_t)(kt + 1) * BK);
      }
    }
    v16h af[4], bf[4];
#pragma unroll
    for (int mi = 0; mi < 4; ++mi) {
      const half_t* pa = &Alds[(wm + mi * 16 + r) * LDP + hk * 8];
      v8h lo = *(const v8h*)pa;
      v8h hi = *(const v8h*)(pa + 16);
      af[mi] = __builtin_shufflevector(lo, hi, 0,1,2,3,4,5,6,7,8,9,10,11,12,13,14,15);
    }
#pragma unroll
    for (int ni = 0; ni < 4; ++ni) {
      const half_t* pb = &Blds[(wn + ni * 16 + r) * LDP + hk * 8];
      v8h lo = *(const v8h*)pb;
      v8h hi = *(const v8h*)(pb + 16);
      bf[ni] = __builtin_shufflevector(lo, hi, 0,1,2,3,4,5,6,7,8,9,10,11,12,13,14,15);
    }
#pragma unroll
    for (int mi = 0; mi < 4; ++mi)
#pragma unroll
      for (int ni = 0; ni < 4; ++ni)
        acc[mi][ni] = __builtin_amdgcn_wmma_f32_16x16x32_f16(
            false, af[mi], false, bf[ni], (short)0, acc[mi][ni], false, false);
  }

  // Scatter C rows to token order. C layout: VGPR rr -> row rr (lanes0-15) /
  // rr+8 (lanes16-31); lane&15 -> column.
  const int ncg = nt * BN + wn + (lane & 15);
#pragma unroll
  for (int mi = 0; mi < 4; ++mi)
#pragma unroll
    for (int ni = 0; ni < 4; ++ni)
#pragma unroll
      for (int rr = 0; rr < 8; ++rr) {
        int ml = wm + mi * 16 + rr + ((lane >> 4) << 3);
        unsigned t = tmap[ml];
        Y[(size_t)t * FF + ncg + ni * 16] = acc[mi][ni][rr];
      }
}

// ---------------------------------------------------------------------------
// K7: out[b][f][s] = Y[b*S+s][f]  (LDS-tiled transpose, coalesced both sides)
// ---------------------------------------------------------------------------
__global__ __launch_bounds__(256) void k_out(const float* __restrict__ Y,
                                             float* __restrict__ out) {
  __shared__ float tile[64 * 65];   // [s][f], pitch 65
  const int b = blockIdx.z, st = blockIdx.y, ft = blockIdx.x;
  const int s0 = st * 64, f0 = ft * 64;
  const int tid = threadIdx.x;
#pragma unroll
  for (int p = 0; p < 4; ++p) {
    int idx = p * 256 + tid;
    int row = idx >> 4, q = idx & 15;
    v4f v = *(const v4f*)(Y + (size_t)(b * SS + s0 + row) * FF + f0 + q * 4);
#pragma unroll
    for (int j = 0; j < 4; ++j) tile[row * 65 + q * 4 + j] = v[j];
  }
  __syncthreads();
#pragma unroll
  for (int p = 0; p < 4; ++p) {
    int idx = p * 256 + tid;
    int ff = idx >> 4, sq = idx & 15;
    v4f v;
#pragma unroll
    for (int j = 0; j < 4; ++j) v[j] = tile[(sq * 4 + j) * 65 + ff];
    *(v4f*)(out + ((size_t)b * FF + f0 + ff) * SS + s0 + sq * 4) = v;
  }
}

// ---------------------------------------------------------------------------
// Host launcher. Workspace layout (bytes):
//   [0,1M)      gatesT   16 x 16384 f32
//   [1M,3M)     keys     16 x 16384 u64
//   [3M,3M+64K) perm     16384 u32
//   [3M+64K,..) scale    16384 f32
//   [4M,68M)    Xs       16384 x 2048 f16
//   [68M,196M)  Wt       16 x 2048 x 2048 f16
//   [196M,324M) Y        16384 x 2048 f32
// ---------------------------------------------------------------------------
extern "C" void kernel_launch(void* const* d_in, const int* in_sizes, int n_in,
                              void* d_out, int out_size, void* d_ws, size_t ws_size,
                              hipStream_t stream) {
  (void)in_sizes; (void)n_in; (void)out_size; (void)ws_size;
  const float* inp  = (const float*)d_in[0];
  const float* W    = (const float*)d_in[1];
  const float* gate = (const float*)d_in[2];
  const int*   fs   = (const int*)d_in[3];
  float* out = (float*)d_out;

  char* ws = (char*)d_ws;
  float*              gatesT = (float*)(ws);
  unsigned long long* keys   = (unsigned long long*)(ws + (1ull << 20));
  unsigned*           perm   = (unsigned*)(ws + (3ull << 20));
  float*              scale  = (float*)(ws + (3ull << 20) + (64ull << 10));
  half_t*             Xs     = (half_t*)(ws + (4ull << 20));
  half_t*             Wt     = (half_t*)(ws + (68ull << 20));
  float*              Y      = (float*)(ws + (196ull << 20));

  k_wt    <<<dim3(32, 32, 16), 256, 0, stream>>>(W, Wt);
  k_gates <<<dim3(64),         256, 0, stream>>>(inp, gate, gatesT);
  k_sort  <<<dim3(16),        1024, 0, stream>>>(gatesT, keys);
  k_select<<<dim3(1),         1024, 0, stream>>>(keys, gatesT, perm, scale);
  k_xs    <<<dim3(64, 32, 8),  256, 0, stream>>>(inp, fs, scale, Xs);
  k_gemm  <<<dim3(16, 8, 16),  128, 0, stream>>>(Xs, Wt, perm, Y);
  k_out   <<<dim3(32, 32, 8),  256, 0, stream>>>(Y, out);
}
